// MultiheadAttention_myversion_16346645528998
// MI455X (gfx1250) — matrix operand order repaired and checked by reading, not compiled, not run on hardware
//
#include <hip/hip_runtime.h>
#include <hip/hip_bf16.h>

// ---------------------------------------------------------------------------
// MultiheadAttention for MI455X (gfx1250), wave32 + WMMA bf16.
// B=2, S=2048, D=1024, H=16, DH=64.
// Outputs: out [B,S,D] fp32 followed by attention_weights [B,H,S,S] fp32.
// ---------------------------------------------------------------------------

#define B_  2
#define S_  2048
#define D_  1024
#define H_  16
#define DH_ 64
#define NEGV (-9.0e15f)

typedef __attribute__((ext_vector_type(16))) __bf16 v16bf;
typedef __attribute__((ext_vector_type(8)))  float  v8f;
typedef __attribute__((ext_vector_type(4)))  int    v4i;

typedef unsigned short u16;

#define AS1 __attribute__((address_space(1)))
#define AS3 __attribute__((address_space(3)))

// Async global->LDS DMA (gfx1250): tracked with ASYNCcnt.  Guarded so the
// build falls back to a synchronous cooperative copy if the builtin is
// absent in this toolchain.  (Probe: param0 is a generic `int4*`, param1 is
// the LDS destination; AS3 -> generic is an implicit superset conversion, so
// casting the destination to AS3 works for either declared signature.)
#if defined(__has_builtin)
#  if __has_builtin(__builtin_amdgcn_global_load_async_to_lds_b128)
#    define HAVE_ASYNC_LDS 1
#  endif
#endif

__device__ __forceinline__ void wait_async_lds() {
#if defined(HAVE_ASYNC_LDS)
#  if __has_builtin(__builtin_amdgcn_s_wait_asynccnt)
    __builtin_amdgcn_s_wait_asynccnt(0);
#  else
    asm volatile("s_wait_asynccnt 0x0" ::: "memory");
#  endif
#endif
}

__device__ __forceinline__ __bf16 f2bf(float f) {
    unsigned int u = __builtin_bit_cast(unsigned int, f);
    u += 0x7FFFu + ((u >> 16) & 1u);                 // round-to-nearest-even
    u16 h = (u16)(u >> 16);
    return __builtin_bit_cast(__bf16, h);
}

__device__ __forceinline__ v8f wmma_bf16(v16bf a, v16bf b, v8f c) {
    return __builtin_amdgcn_wmma_f32_16x16x32_bf16(
        false, a, false, b, (short)0, c, false, false);
}

// A-matrix 16x32 bf16 fragment (contiguous k order used here, same as load_b16): lane holds its row r=lane&15.
// With hi=lane>>4, elements e=0..15 are K = k0+hi*16+e.
// `p` must point at X[row*ld + k0 + hi*16].
__device__ __forceinline__ v16bf load_a16(const float* p) {
    v16bf a;
#pragma unroll
    for (int e = 0; e < 8; ++e) a[e] = f2bf(p[e]);
#pragma unroll
    for (int e = 0; e < 8; ++e) a[8 + e] = f2bf(p[8 + e]);
    return a;
}
__device__ __forceinline__ v16bf load_a16(const u16* p) {
    v16bf a;
#pragma unroll
    for (int e = 0; e < 8; ++e) a[e] = __builtin_bit_cast(__bf16, p[e]);
#pragma unroll
    for (int e = 0; e < 8; ++e) a[8 + e] = __builtin_bit_cast(__bf16, p[8 + e]);
    return a;
}

// B-matrix 32x16 bf16 fragment: lane holds column n=lane&15, which for
// B = W^T (or K^T) is a contiguous row of the row-major source.  Elements
// e=0..15 hold K = k0 + hi*16 + e.  `p` points at Src[n*ld + k0 + hi*16].
__device__ __forceinline__ v16bf load_b16(const float* p) {
    v16bf b;
#pragma unroll
    for (int e = 0; e < 16; ++e) b[e] = f2bf(p[e]);
    return b;
}
__device__ __forceinline__ v16bf load_b16(const u16* p) {
    v16bf b;
#pragma unroll
    for (int e = 0; e < 16; ++e) b[e] = __builtin_bit_cast(__bf16, p[e]);
    return b;
}

__device__ __forceinline__ v8f vzero8() {
    v8f c;
#pragma unroll
    for (int i = 0; i < 8; ++i) c[i] = 0.0f;
    return c;
}

// ---------------------------------------------------------------------------
// Generic GEMM:  Y[M,N] = X[M,K] @ W[N,K]^T + bias[N]
//   T = float  : X fp32, converted to bf16 on the fly
//   T = u16    : X already bf16
//   HEAD_OUT=1 : Y written bf16, scattered to [B,H,S,DH] head-major layout
//   HEAD_OUT=0 : Y written fp32 row-major [M,N]
// Block: 256 threads (8 waves), wave grid 2(M) x 4(N), 64x64 C tile per wave.
// Workgroup tile: 128(M) x 256(N).
// ---------------------------------------------------------------------------
template <typename T, bool HEAD_OUT>
__global__ __launch_bounds__(256)
void gemm_bias(const T* __restrict__ X, const float* __restrict__ W,
               const float* __restrict__ bias, void* __restrict__ Yv,
               int M, int N, int K) {
    const int tid  = threadIdx.x;
    const int wave = tid >> 5;
    const int lane = tid & 31;
    const int r    = lane & 15;
    const int hi   = lane >> 4;

    const int wm = wave >> 2;          // 0..1
    const int wn = wave & 3;           // 0..3
    const int mbase = blockIdx.y * 128 + wm * 64;
    const int nbase = blockIdx.x * 256 + wn * 64;

    v8f c[4][4];
#pragma unroll
    for (int mt = 0; mt < 4; ++mt)
#pragma unroll
        for (int nt = 0; nt < 4; ++nt) c[mt][nt] = vzero8();

    for (int k0 = 0; k0 < K; k0 += 32) {
        v16bf a[4];
#pragma unroll
        for (int mt = 0; mt < 4; ++mt) {
            const T* pa = X + (size_t)(mbase + mt * 16 + r) * K + k0 + hi * 16;
            a[mt] = load_a16(pa);
            __builtin_prefetch(pa + 32, 0, 0);   // next k-slice of X
        }
#pragma unroll
        for (int nt = 0; nt < 4; ++nt) {
            const float* pb = W + (size_t)(nbase + nt * 16 + r) * K + k0 + hi * 16;
            v16bf b = load_b16(pb);
            __builtin_prefetch(pb + 32, 0, 0);   // next k-slice of W
#pragma unroll
            for (int mt = 0; mt < 4; ++mt)
                c[mt][nt] = wmma_bf16(a[mt], b, c[mt][nt]);
        }
    }

    // C layout: element v -> row M = v + hi*8, col N = lane&15
#pragma unroll
    for (int mt = 0; mt < 4; ++mt) {
#pragma unroll
        for (int nt = 0; nt < 4; ++nt) {
            const int ncol = nbase + nt * 16 + r;
            const float bb = bias[ncol];
#pragma unroll
            for (int v = 0; v < 8; ++v) {
                const int mrow = mbase + mt * 16 + v + hi * 8;
                const float val = c[mt][nt][v] + bb;
                if (HEAD_OUT) {
                    u16* Y = (u16*)Yv;
                    const int b  = mrow / S_;
                    const int s  = mrow - b * S_;
                    const int h  = ncol >> 6;
                    const int dh = ncol & 63;
                    Y[(((size_t)b * H_ + h) * S_ + s) * DH_ + dh] =
                        __builtin_bit_cast(u16, f2bf(val));
                } else {
                    float* Y = (float*)Yv;
                    Y[(size_t)mrow * N + ncol] = val;
                }
            }
        }
    }
}

// ---------------------------------------------------------------------------
// Attention: one block = (b, h, 32 query rows).  128 threads = 4 waves.
// LDS: 32 x (2048+8) fp32 logits (257 KB) + double-buffered 32x64 bf16 V
// chunk (8 KB) -> 271,360 B of the WGP's 320 KB.
// Phase 1: logits = Q K^T / 8 + masks (WMMA)
// Phase 2: row softmax, weights -> d_out (coalesced fp32)
// Phase 3: O = P V (WMMA), V streamed via async global->LDS DMA,
//          O -> bf16 [B,S,D] activation buffer
// ---------------------------------------------------------------------------
#define LDSW   (S_ + 8)
#define VCHUNK (32 * DH_)                       // u16 elements per V chunk
#define SMEM_BYTES ((size_t)32 * LDSW * sizeof(float) + 2 * VCHUNK * sizeof(u16))

// Cooperatively move one 32-key x 64-dh bf16 V chunk (4 KB) into LDS.
// 256 x 16B transfers, 2 per thread.
__device__ __forceinline__ void copy_v_chunk(const u16* __restrict__ Vp,
                                             u16* __restrict__ dst,
                                             int kb, int tid) {
#pragma unroll
    for (int i = 0; i < 2; ++i) {
        const int idx = tid + i * 128;          // 0..255
        const int row = idx >> 3;
        const int seg = idx & 7;
        const u16* src = Vp + (size_t)(kb + row) * DH_ + seg * 8;
        u16* d = dst + row * DH_ + seg * 8;
#if defined(HAVE_ASYNC_LDS)
        __builtin_amdgcn_global_load_async_to_lds_b128(
            (v4i*)src, (AS3 v4i*)d, 0, 0);
#else
        *(uint4*)d = *(const uint4*)src;
#endif
    }
}

__global__ __launch_bounds__(128)
void attn_kernel(const u16* __restrict__ Qb, const u16* __restrict__ Kb,
                 const u16* __restrict__ Vb,
                 const unsigned char* __restrict__ kpad,
                 float* __restrict__ attw, u16* __restrict__ Xattn) {
    extern __shared__ char smraw[];
    float* sm  = (float*)smraw;                                  // [32][LDSW]
    u16*  vbuf = (u16*)(smraw + (size_t)32 * LDSW * sizeof(float)); // [2][VCHUNK]

    const int qt = blockIdx.x;       // 0..63
    const int h  = blockIdx.y;       // 0..15
    const int b  = blockIdx.z;       // 0..1
    const size_t bh = (size_t)b * H_ + h;

    const u16* Qp = Qb + bh * S_ * DH_;
    const u16* Kp = Kb + bh * S_ * DH_;
    const u16* Vp = Vb + bh * S_ * DH_;

    const int tid  = threadIdx.x;
    const int wave = tid >> 5;
    const int lane = tid & 31;
    const int r    = lane & 15;
    const int hi   = lane >> 4;

    const int qhalf = wave >> 1;                 // 16-row half owned by wave
    const int qbase = qt * 32 + qhalf * 16;

    // Q fragments for this wave's 16 rows, dh chunks [0,32) and [32,64)
    const u16* qrow = Qp + (size_t)(qbase + r) * DH_ + hi * 16;
    v16bf aq0 = load_a16(qrow);
    v16bf aq1 = load_a16(qrow + 32);

    // ---- Phase 1: logits tiles ------------------------------------------
    for (int kt = (wave & 1); kt < S_ / 16; kt += 2) {
        const int kb = kt * 16;
        v8f c = vzero8();
        {
            const u16* pk = Kp + (size_t)(kb + r) * DH_ + hi * 16;
            if (kt + 2 < S_ / 16)
                __builtin_prefetch(pk + 32 * DH_, 0, 0);  // K rows of tile kt+2
            c = wmma_bf16(aq0, load_b16(pk), c);
            c = wmma_bf16(aq1, load_b16(pk + 32), c);
        }
        const int key = kb + r;
        const bool pad = (kpad[(size_t)b * S_ + key] != 0);
#pragma unroll
        for (int v = 0; v < 8; ++v) {
            const int qloc = qhalf * 16 + v + hi * 8;   // 0..31
            const int qg   = qt * 32 + qloc;
            const bool msk = pad || (key > qg);          // causal triu(k=1)
            sm[(size_t)qloc * LDSW + key] = msk ? NEGV : c[v] * 0.125f;
        }
    }
    __syncthreads();

    // ---- Phase 2: softmax over full rows, write attention weights -------
    for (int rr = 0; rr < 8; ++rr) {
        const int row = wave * 8 + rr;
        float* Lr = sm + (size_t)row * LDSW;
        float mx = -3.0e38f;
        for (int j = lane; j < S_; j += 32) mx = fmaxf(mx, Lr[j]);
#pragma unroll
        for (int off = 16; off > 0; off >>= 1)
            mx = fmaxf(mx, __shfl_xor(mx, off, 32));
        float sum = 0.0f;
        for (int j = lane; j < S_; j += 32) {
            const float e = __expf(Lr[j] - mx);
            Lr[j] = e;                       // cache unnormalized exp
            sum += e;
        }
#pragma unroll
        for (int off = 16; off > 0; off >>= 1)
            sum += __shfl_xor(sum, off, 32);
        const float inv = 1.0f / sum;
        const int qg = qt * 32 + row;
        float* wout = attw + (bh * S_ + qg) * (size_t)S_;
        for (int j = lane; j < S_; j += 32) {
            const float p = Lr[j] * inv;
            Lr[j]   = p;          // normalized probs for P@V
            wout[j] = p;          // coalesced global write
        }
    }
    __syncthreads();

    // ---- Phase 3: O = P @ V, V double-buffered through LDS --------------
    const int dhbase = (wave & 1) * 32;
    v8f o0 = vzero8(), o1 = vzero8();

    copy_v_chunk(Vp, vbuf, 0, tid);                  // prime buffer 0
    for (int kc = 0; kc < S_ / 32; ++kc) {
        wait_async_lds();
        __syncthreads();                             // chunk kc resident
        if (kc + 1 < S_ / 32)
            copy_v_chunk(Vp, vbuf + ((kc + 1) & 1) * VCHUNK, (kc + 1) * 32, tid);

        const int kb = kc * 32;
        const u16* vc = vbuf + (kc & 1) * VCHUNK;    // [32 keys][64 dh]
        const float* pp = sm + (size_t)(qhalf * 16 + r) * LDSW + kb + hi * 16;
        v16bf ap = load_a16(pp);                     // probs, f32 -> bf16
#pragma unroll
        for (int nt = 0; nt < 2; ++nt) {
            const int dh0 = dhbase + nt * 16;
            // B = V chunk [32 x 16]: lane holds column n=r, K = e + hi*16
            const u16* vpl = vc + (size_t)(hi * 16) * DH_ + dh0 + r;
            v16bf bv;
#pragma unroll
            for (int e = 0; e < 16; ++e)
                bv[e] = __builtin_bit_cast(__bf16, vpl[(size_t)e * DH_]);
            if (nt == 0) o0 = wmma_bf16(ap, bv, o0);
            else         o1 = wmma_bf16(ap, bv, o1);
        }
    }
#pragma unroll
    for (int nt = 0; nt < 2; ++nt) {
        const v8f& o = nt ? o1 : o0;
        const int col = h * DH_ + dhbase + nt * 16 + r;  // column in [0,1024)
#pragma unroll
        for (int v = 0; v < 8; ++v) {
            const int qg = qt * 32 + qhalf * 16 + v + hi * 8;
            Xattn[((size_t)b * S_ + qg) * D_ + col] =
                __builtin_bit_cast(u16, f2bf(o[v]));
        }
    }
}

// ---------------------------------------------------------------------------
extern "C" void kernel_launch(void* const* d_in, const int* in_sizes, int n_in,
                              void* d_out, int out_size, void* d_ws, size_t ws_size,
                              hipStream_t stream) {
    (void)in_sizes; (void)n_in; (void)out_size; (void)ws_size;

    const float* q    = (const float*)d_in[0];
    const float* k    = (const float*)d_in[1];
    const float* v    = (const float*)d_in[2];
    const float* Wq_w = (const float*)d_in[3];
    const float* Wq_b = (const float*)d_in[4];
    const float* Wk_w = (const float*)d_in[5];
    const float* Wk_b = (const float*)d_in[6];
    const float* Wv_w = (const float*)d_in[7];
    const float* Wv_b = (const float*)d_in[8];
    const float* fc_w = (const float*)d_in[9];
    const float* fc_b = (const float*)d_in[10];
    // d_in[11] = attn_mask (triu k=1, computed analytically in-kernel)
    const unsigned char* kpad = (const unsigned char*)d_in[12];

    float* out  = (float*)d_out;                               // [B,S,D]
    float* attw = out + (size_t)B_ * S_ * D_;                  // [B,H,S,S]

    const size_t nElem = (size_t)B_ * S_ * D_;                 // 4 Mi
    u16* Qb    = (u16*)d_ws;
    u16* Kb    = Qb + nElem;
    u16* Vb    = Kb + nElem;
    u16* Xattn = Vb + nElem;                                   // bf16 [B,S,D]

    const int M = B_ * S_, N = D_, K = D_;
    dim3 gblk(256);
    dim3 ggrid(N / 256, M / 128);

    // Q/K/V projections -> bf16 head-major
    gemm_bias<float, true><<<ggrid, gblk, 0, stream>>>(q, Wq_w, Wq_b, Qb, M, N, K);
    gemm_bias<float, true><<<ggrid, gblk, 0, stream>>>(k, Wk_w, Wk_b, Kb, M, N, K);
    gemm_bias<float, true><<<ggrid, gblk, 0, stream>>>(v, Wv_w, Wv_b, Vb, M, N, K);

    // Attention: ~271 KB dynamic LDS per workgroup (WGP has 320 KB)
    (void)hipFuncSetAttribute((const void*)attn_kernel,
                              hipFuncAttributeMaxDynamicSharedMemorySize,
                              (int)SMEM_BYTES);
    dim3 agrid(S_ / 32, H_, B_);
    attn_kernel<<<agrid, dim3(128), SMEM_BYTES, stream>>>(Qb, Kb, Vb, kpad,
                                                          attw, Xattn);

    // Final projection -> fp32 out
    gemm_bias<u16, false><<<ggrid, gblk, 0, stream>>>(Xattn, fc_w, fc_b, out, M, N, K);
}
